// AttenGcnLayer_77799037600427
// MI455X (gfx1250) — matrix operand order repaired and checked by reading, not compiled, not run on hardware
//
#include <hip/hip_runtime.h>

#define HID      128
#define N_NODES  50000
#define N_REL    500
#define N_EDGES  625000
#define WAVES    4           // waves per block (block = 128 threads, wave32)
#define LRELU    0.01f

typedef _Float16 half_t;
typedef half_t v16h __attribute__((ext_vector_type(16)));
typedef float  v8f  __attribute__((ext_vector_type(8)));

// ---------------------------------------------------------------------------
// helpers
// ---------------------------------------------------------------------------

// LDS is serviced in-order per wave (DScnt); this enforces store->load visibility
// across lanes of the SAME wave without a block barrier.
static __device__ __forceinline__ void wave_lds_fence() {
  __asm__ volatile("s_wait_dscnt 0" ::: "memory");
}

// float atomic max via monotone int/uint encoding (seg_max initialized to -inf)
static __device__ __forceinline__ void atomicMaxF(float* addr, float v) {
  if (v >= 0.f) atomicMax((int*)addr, __float_as_int(v));
  else          atomicMin((unsigned int*)addr, __float_as_uint(v));
}

// A fragment (16x32 f16, MxK) from a row-major LDS tile.
// lane<16: M=lane, K = kb+0..7 (v0..3) and kb+16..23 (v4..7); lane>=16: kb += 8.
static __device__ __forceinline__ v16h load_A_frag(const half_t* tile, int ldk,
                                                   int kt, int lane) {
  const int m  = lane & 15;
  const int kb = kt * 32 + ((lane & 16) ? 8 : 0);
  const half_t* rp = tile + m * ldk + kb;
  v16h a;
#pragma unroll
  for (int i = 0; i < 8; ++i) { a[i] = rp[i]; a[8 + i] = rp[16 + i]; }
  return a;
}

// B fragment: 8 pre-packed dwords per lane (two b128 loads).
static __device__ __forceinline__ v16h load_B_frag(const unsigned int* p) {
  union { unsigned int u[8]; v16h v; } c;
  const uint4* q = (const uint4*)p;
  uint4 a = q[0], b = q[1];
  c.u[0] = a.x; c.u[1] = a.y; c.u[2] = a.z; c.u[3] = a.w;
  c.u[4] = b.x; c.u[5] = b.y; c.u[6] = b.z; c.u[7] = b.w;
  return c.v;
}

// 16x128 = (16x256 f16) x (256x128 f16) with f32 accumulate: 64 WMMAs.
static __device__ __forceinline__ void mess_tile(const half_t* ldsA,
                                                 const unsigned int* Bm,
                                                 int lane, v8f acc[8]) {
  v8f z = {};
#pragma unroll
  for (int nt = 0; nt < 8; ++nt) acc[nt] = z;
#pragma unroll
  for (int kt = 0; kt < 8; ++kt) {
    v16h a = load_A_frag(ldsA, 256, kt, lane);
#pragma unroll
    for (int nt = 0; nt < 8; ++nt) {
      v16h b = load_B_frag(Bm + (((kt << 3) + nt) * 32 + lane) * 8);
      acc[nt] = __builtin_amdgcn_wmma_f32_16x16x32_f16(
          false, a, false, b, (short)0, acc[nt], false, false);
    }
  }
}

// Gather 16 edges' [head_embed | rel_embed] rows (f32 -> f16) into a wave-private
// LDS tile 16x256; also stash the 16 tail indices.
static __device__ __forceinline__ void gather_edges(
    const float* __restrict__ x, const float* __restrict__ rbn,
    const int* __restrict__ heads, const int* __restrict__ rels,
    const int* __restrict__ tails, long e0, int lane,
    half_t (*ldsA)[256], int* ldsTail) {
  if (lane < 16) {
    long e = e0 + lane; if (e >= (long)N_EDGES) e = N_EDGES - 1;
    ldsTail[lane] = tails[e];
  }
  const int row = lane >> 1;
  const int hf  = (lane & 1) * 64;
  long e = e0 + row; if (e >= (long)N_EDGES) e = N_EDGES - 1;
  const float* hs = x   + (long)heads[e] * HID + hf;
  const float* rs = rbn + (long)rels[e]  * HID + hf;
  half_t* dh = &ldsA[row][hf];
  half_t* dr = &ldsA[row][HID + hf];
#pragma unroll 4
  for (int i = 0; i < 64; i += 4) {
    float4 a = *(const float4*)(hs + i);
    dh[i] = (half_t)a.x; dh[i+1] = (half_t)a.y; dh[i+2] = (half_t)a.z; dh[i+3] = (half_t)a.w;
    float4 b = *(const float4*)(rs + i);
    dr[i] = (half_t)b.x; dr[i+1] = (half_t)b.y; dr[i+2] = (half_t)b.z; dr[i+3] = (half_t)b.w;
  }
}

// ---------------------------------------------------------------------------
// small setup kernels
// ---------------------------------------------------------------------------

__global__ void k_init(float* __restrict__ sum_mess, float* __restrict__ seg_max,
                       float* __restrict__ seg_sum) {
  long i = (long)blockIdx.x * 256 + threadIdx.x;
  if (i < (long)N_NODES * HID) sum_mess[i] = 0.f;
  if (i < N_NODES) { seg_max[i] = -__builtin_huge_valf(); seg_sum[i] = 0.f; }
}

// BatchNorm1d over r (training mode, biased variance). 1 block x 128 threads.
__global__ void k_bn(const float* __restrict__ r, const float* __restrict__ gamma,
                     const float* __restrict__ beta, float* __restrict__ r_out) {
  const int f = threadIdx.x;
  float s = 0.f;
  for (int i = 0; i < N_REL; ++i) s += r[i * HID + f];
  const float mu = s * (1.f / N_REL);
  float v = 0.f;
  for (int i = 0; i < N_REL; ++i) { float d = r[i * HID + f] - mu; v += d * d; }
  v *= (1.f / N_REL);
  const float inv = rsqrtf(v + 1e-5f) * gamma[f];
  const float b = beta[f];
  for (int i = 0; i < N_REL; ++i)
    r_out[i * HID + f] = (r[i * HID + f] - mu) * inv + b;
}

// cq[h] = b_matt[h] + W_matt[h,128:] . que   ;  cf[h] = b_xatt[h] + W_xatt[h,128:] . fin_que
__global__ void k_prep(const float* __restrict__ W_matt, const float* __restrict__ b_matt,
                       const float* __restrict__ que,
                       const float* __restrict__ W_xatt, const float* __restrict__ b_xatt,
                       const float* __restrict__ finq,
                       float* __restrict__ cq, float* __restrict__ cf) {
  const int t = threadIdx.x;
  if (t < HID) {
    float s = b_matt[t];
    for (int c = 0; c < HID; ++c) s += W_matt[t * 2 * HID + HID + c] * que[c];
    cq[t] = s;
  } else {
    const int f = t - HID;
    float s = b_xatt[f];
    for (int c = 0; c < HID; ++c) s += W_xatt[f * 2 * HID + HID + c] * finq[c];
    cf[f] = s;
  }
}

// Pack B operands into per-lane WMMA layout (f16 pairs per dword).
// B is KxN with B[k,n] = W[n,k]. lane<16: K=2v,2v+1 (+8 for v>=4) (n=lane); lane>=16: K=8+2v (+8 for v>=4) (n=lane-16).
// Region: [0,16384) W_mess (kt 0..7) | [16384,24576) W_matt[:, :128] | [24576,32768) W_xatt[:, :128]
__global__ void k_pack(const float* __restrict__ Wm, const float* __restrict__ Wmatt,
                       const float* __restrict__ Wx, unsigned int* __restrict__ pack) {
  const int idx = blockIdx.x * 256 + threadIdx.x;
  if (idx >= 32768) return;
  const float* W; int rel;
  if (idx < 16384)      { W = Wm;    rel = idx; }
  else if (idx < 24576) { W = Wmatt; rel = idx - 16384; }
  else                  { W = Wx;    rel = idx - 24576; }
  const int kt   = rel >> 11;
  const int nt   = (rel >> 8) & 7;
  const int lane = (rel >> 3) & 31;
  const int v    = rel & 7;
  const int k0 = kt * 32 + ((lane & 16) ? 8 : 0) + 2 * v + ((v >= 4) ? 8 : 0);
  const int n  = nt * 16 + (lane & 15);
  union { half_t h[2]; unsigned int u; } c;
  c.h[0] = (half_t)W[n * 256 + k0];
  c.h[1] = (half_t)W[n * 256 + k0 + 1];
  pack[idx] = c.u;
}

// ---------------------------------------------------------------------------
// edge pass 1: mess GEMM (K=256) -> tanh -> attention GEMM (K=128) -> logits,
// atomic segment max over tails, store logits.
// ---------------------------------------------------------------------------
__global__ __launch_bounds__(128) void k_edge_pass1(
    const float* __restrict__ x, const float* __restrict__ rbn,
    const int* __restrict__ heads, const int* __restrict__ tails,
    const int* __restrict__ rels,
    const unsigned int* __restrict__ Bm, const unsigned int* __restrict__ Bmatt,
    const float* __restrict__ b_mess, const float* __restrict__ cq,
    const float* __restrict__ maw,
    float* __restrict__ coeff, float* __restrict__ seg_max) {
  __shared__ half_t ldsA[WAVES][16][256];
  __shared__ half_t ldsM[WAVES][16][128];
  __shared__ int    ldsTail[WAVES][16];
  const int wave = threadIdx.x >> 5, lane = threadIdx.x & 31;
  const long e0 = ((long)blockIdx.x * WAVES + wave) * 16;
  if (e0 >= (long)N_EDGES) return;

  gather_edges(x, rbn, heads, rels, tails, e0, lane, ldsA[wave], ldsTail[wave]);
  wave_lds_fence();

  v8f acc[8];
  mess_tile(&ldsA[wave][0][0], Bm, lane, acc);

  const int hi  = (lane & 16) ? 8 : 0;
  const int nlo = lane & 15;
#pragma unroll
  for (int nt = 0; nt < 8; ++nt) {
    const int n = nt * 16 + nlo;
    const float bn = b_mess[n];
#pragma unroll
    for (int i = 0; i < 8; ++i)
      ldsM[wave][i + hi][n] = (half_t)tanhf(acc[nt][i] + bn);
  }
  wave_lds_fence();

  // mc = lrelu(mess @ Wmatt1^T + cq), logits = maw . mc
  v8f acc2[8];
#pragma unroll
  for (int nt = 0; nt < 8; ++nt) {
    const float c = cq[nt * 16 + nlo];
#pragma unroll
    for (int i = 0; i < 8; ++i) acc2[nt][i] = c;
  }
#pragma unroll
  for (int kt = 0; kt < 4; ++kt) {
    v16h a = load_A_frag(&ldsM[wave][0][0], 128, kt, lane);
#pragma unroll
    for (int nt = 0; nt < 8; ++nt) {
      v16h b = load_B_frag(Bmatt + (((kt << 3) + nt) * 32 + lane) * 8);
      acc2[nt] = __builtin_amdgcn_wmma_f32_16x16x32_f16(
          false, a, false, b, (short)0, acc2[nt], false, false);
    }
  }
  float p[8] = {0, 0, 0, 0, 0, 0, 0, 0};
#pragma unroll
  for (int nt = 0; nt < 8; ++nt) {
    const float w = maw[nt * 16 + nlo];
#pragma unroll
    for (int i = 0; i < 8; ++i) {
      float v = acc2[nt][i];
      v = v > 0.f ? v : LRELU * v;
      p[i] += w * v;
    }
  }
#pragma unroll
  for (int m = 1; m < 16; m <<= 1)
#pragma unroll
    for (int i = 0; i < 8; ++i) p[i] += __shfl_xor(p[i], m, 32);

  if (nlo == 0) {
#pragma unroll
    for (int i = 0; i < 8; ++i) {
      const int row = i + hi;
      const long e = e0 + row;
      if (e < (long)N_EDGES) {
        coeff[e] = p[i];
        atomicMaxF(seg_max + ldsTail[wave][row], p[i]);
      }
    }
  }
}

// edge pass 2a: w = exp(logit - segmax[tail]); segment sums
__global__ void k_edge_pass2a(const int* __restrict__ tails,
                              const float* __restrict__ seg_max,
                              float* __restrict__ coeff,
                              float* __restrict__ seg_sum) {
  const long e = (long)blockIdx.x * 256 + threadIdx.x;
  if (e >= (long)N_EDGES) return;
  const int t = tails[e];
  const float w = __expf(coeff[e] - seg_max[t]);
  coeff[e] = w;
  atomicAdd(seg_sum + t, w);
}

// edge pass 2b: recompute mess tile (cheaper than spilling 160MB), normalize,
// scatter-add into sum_mess (L2-resident global_atomic_add_f32).
__global__ __launch_bounds__(128) void k_edge_pass2b(
    const float* __restrict__ x, const float* __restrict__ rbn,
    const int* __restrict__ heads, const int* __restrict__ tails,
    const int* __restrict__ rels,
    const unsigned int* __restrict__ Bm, const float* __restrict__ b_mess,
    const float* __restrict__ wexp, const float* __restrict__ seg_sum,
    float* __restrict__ sum_mess) {
  __shared__ half_t ldsA[WAVES][16][256];
  __shared__ int    ldsTail[WAVES][16];
  const int wave = threadIdx.x >> 5, lane = threadIdx.x & 31;
  const long e0 = ((long)blockIdx.x * WAVES + wave) * 16;
  if (e0 >= (long)N_EDGES) return;

  gather_edges(x, rbn, heads, rels, tails, e0, lane, ldsA[wave], ldsTail[wave]);
  wave_lds_fence();

  v8f acc[8];
  mess_tile(&ldsA[wave][0][0], Bm, lane, acc);

  const int hi  = (lane & 16) ? 8 : 0;
  const int nlo = lane & 15;
  float wn[8]; int tl[8];
#pragma unroll
  for (int i = 0; i < 8; ++i) {
    const long e = e0 + i + hi;
    if (e < (long)N_EDGES) {
      tl[i] = ldsTail[wave][i + hi];
      wn[i] = wexp[e] / (seg_sum[tl[i]] + 1e-16f);
    } else { tl[i] = 0; wn[i] = 0.f; }
  }
#pragma unroll
  for (int nt = 0; nt < 8; ++nt) {
    const int n = nt * 16 + nlo;
    const float bn = b_mess[n];
#pragma unroll
    for (int i = 0; i < 8; ++i) {
      if (wn[i] != 0.f) {
        const float mv = tanhf(acc[nt][i] + bn);
        atomicAdd(sum_mess + (long)tl[i] * HID + n, mv * wn[i]);
      }
    }
  }
}

// node gating: rows interleaved [x[node]; sum_mess[node]] through W_xatt1,
// 2-way softmax per node, blend. sum_mess lives in d_out and is overwritten
// in place (each node owned by exactly one wave; reads precede writes).
__global__ __launch_bounds__(128) void k_node_gate(
    const float* __restrict__ x, float* __restrict__ xout,
    const unsigned int* __restrict__ Bx, const float* __restrict__ cf,
    const float* __restrict__ xaw) {
  __shared__ half_t ldsA[WAVES][16][128];
  const int wave = threadIdx.x >> 5, lane = threadIdx.x & 31;
  const long node0 = ((long)blockIdx.x * WAVES + wave) * 8;
  if (node0 >= (long)N_NODES) return;
  {
    const int row = lane >> 1;
    const int hf  = (lane & 1) * 64;
    const long node = node0 + (row >> 1);
    const float* src = ((row & 1) ? xout : x) + node * HID + hf;
    half_t* d = &ldsA[wave][row][hf];
#pragma unroll 4
    for (int i = 0; i < 64; i += 4) {
      float4 a = *(const float4*)(src + i);
      d[i] = (half_t)a.x; d[i+1] = (half_t)a.y; d[i+2] = (half_t)a.z; d[i+3] = (half_t)a.w;
    }
  }
  wave_lds_fence();

  const int hi  = (lane & 16) ? 8 : 0;
  const int nlo = lane & 15;
  v8f acc[8];
#pragma unroll
  for (int nt = 0; nt < 8; ++nt) {
    const float c = cf[nt * 16 + nlo];
#pragma unroll
    for (int i = 0; i < 8; ++i) acc[nt][i] = c;
  }
#pragma unroll
  for (int kt = 0; kt < 4; ++kt) {
    v16h a = load_A_frag(&ldsA[wave][0][0], 128, kt, lane);
#pragma unroll
    for (int nt = 0; nt < 8; ++nt) {
      v16h b = load_B_frag(Bx + (((kt << 3) + nt) * 32 + lane) * 8);
      acc[nt] = __builtin_amdgcn_wmma_f32_16x16x32_f16(
          false, a, false, b, (short)0, acc[nt], false, false);
    }
  }
  float p[8] = {0, 0, 0, 0, 0, 0, 0, 0};
#pragma unroll
  for (int nt = 0; nt < 8; ++nt) {
    const float w = xaw[nt * 16 + nlo];
#pragma unroll
    for (int i = 0; i < 8; ++i) {
      float v = acc[nt][i];
      v = v > 0.f ? v : LRELU * v;
      p[i] += w * v;
    }
  }
#pragma unroll
  for (int m = 1; m < 16; m <<= 1)
#pragma unroll
    for (int i = 0; i < 8; ++i) p[i] += __shfl_xor(p[i], m, 32);

  const long nbase = node0 + (hi ? 4 : 0);
#pragma unroll
  for (int j = 0; j < 4; ++j) {
    const float a0 = p[2 * j], a1 = p[2 * j + 1];
    const float mx = fmaxf(a0, a1);
    const float ea = __expf(a0 - mx), eb = __expf(a1 - mx);
    const float inv = 1.f / (ea + eb);
    const float w0 = ea * inv, w1 = eb * inv;
    const long node = nbase + j;
    for (int f = nlo; f < HID; f += 16) {
      const long idx = node * HID + f;
      const float xv = x[idx], sv = xout[idx];
      xout[idx] = w0 * xv + w1 * sv;
    }
  }
}

// ---------------------------------------------------------------------------
extern "C" void kernel_launch(void* const* d_in, const int* in_sizes, int n_in,
                              void* d_out, int out_size, void* d_ws, size_t ws_size,
                              hipStream_t stream) {
  const float* x      = (const float*)d_in[0];
  const float* r      = (const float*)d_in[1];
  const float* que    = (const float*)d_in[2];
  const float* finq   = (const float*)d_in[3];
  const int*   eidx   = (const int*)d_in[4];
  const int*   eattr  = (const int*)d_in[5];
  const float* W_mess = (const float*)d_in[6];
  const float* b_mess = (const float*)d_in[7];
  const float* W_matt = (const float*)d_in[8];
  const float* b_matt = (const float*)d_in[9];
  const float* maw    = (const float*)d_in[10];
  const float* W_xatt = (const float*)d_in[11];
  const float* b_xatt = (const float*)d_in[12];
  const float* xaw    = (const float*)d_in[13];
  const float* gamma  = (const float*)d_in[14];
  const float* beta   = (const float*)d_in[15];

  float* x_out = (float*)d_out;                       // also sum_mess staging
  float* r_out = (float*)d_out + (long)N_NODES * HID; // r_bn output

  // workspace layout (≈ 3 MB)
  unsigned int* pack = (unsigned int*)d_ws;  // 32768 dwords
  unsigned int* Bm    = pack;                // W_mess   (8 ktiles)
  unsigned int* Bmatt = pack + 16384;        // W_matt[:, :128] (4 ktiles)
  unsigned int* Bx    = pack + 24576;        // W_xatt[:, :128] (4 ktiles)
  float* cq      = (float*)(pack + 32768);
  float* cf      = cq + HID;
  float* seg_max = cf + HID;
  float* seg_sum = seg_max + N_NODES;
  float* coeff   = seg_sum + N_NODES;        // E floats (logits, then exp weights)

  const int* heads = eidx;
  const int* tails = eidx + N_EDGES;
  const int* rels  = eattr;

  const long NH = (long)N_NODES * HID;
  k_init<<<(int)((NH + 255) / 256), 256, 0, stream>>>(x_out, seg_max, seg_sum);
  k_bn<<<1, 128, 0, stream>>>(r, gamma, beta, r_out);
  k_prep<<<1, 256, 0, stream>>>(W_matt, b_matt, que, W_xatt, b_xatt, finq, cq, cf);
  k_pack<<<(32768 + 255) / 256, 256, 0, stream>>>(W_mess, W_matt, W_xatt, pack);

  const int nTiles  = (N_EDGES + 15) / 16;
  const int eBlocks = (nTiles + WAVES - 1) / WAVES;
  k_edge_pass1<<<eBlocks, 128, 0, stream>>>(x, r_out, heads, tails, rels,
                                            Bm, Bmatt, b_mess, cq, maw,
                                            coeff, seg_max);
  k_edge_pass2a<<<(N_EDGES + 255) / 256, 256, 0, stream>>>(tails, seg_max, coeff, seg_sum);
  k_edge_pass2b<<<eBlocks, 128, 0, stream>>>(x, r_out, heads, tails, rels,
                                             Bm, b_mess, coeff, seg_sum, x_out);

  const int nodeTiles = (N_NODES + 7) / 8;
  k_node_gate<<<(nodeTiles + WAVES - 1) / WAVES, 128, 0, stream>>>(x, x_out, Bx, cf, xaw);
}
